// AttentionLayer_56994216018002
// MI455X (gfx1250) — matrix operand order repaired and checked by reading, not compiled, not run on hardware
//
#include <hip/hip_runtime.h>
#include <hip/hip_bf16.h>
#include <math.h>

// ---------------------------------------------------------------------------
// Types for CDNA5 WMMA (wave32): bf16 16x16x32, f32 accumulate
// ---------------------------------------------------------------------------
typedef __bf16 bf16;
typedef __attribute__((ext_vector_type(16))) __bf16 v16bf;
typedef __attribute__((ext_vector_type(8)))  __bf16 v8bf;
typedef __attribute__((ext_vector_type(4)))  __bf16 v4bf;
typedef __attribute__((ext_vector_type(8)))  float  v8f;

#define HIDDEN   3072
#define HEADS    32
#define HDIM     96
#define BATCH    2
#define SEQ      2048
#define ROWS     (BATCH*SEQ)     // 4096 tokens
#define NQKV     (3*HIDDEN)      // 9216
#define NBH      (BATCH*HEADS)   // 64
#define QK_SCALE 0.10206207262f  // 1/sqrt(96)

__device__ __forceinline__ v8f wmma_bf16(v16bf a, v16bf b, v8f c) {
  // 8 args: (neg_a, A, neg_b, B, c_mod, C, reuse_a, reuse_b)
  return __builtin_amdgcn_wmma_f32_16x16x32_bf16(false, a, false, b, (short)0, c,
                                                 false, false);
}
__device__ __forceinline__ v16bf cat8(v8bf lo, v8bf hi) {
  return __builtin_shufflevector(lo, hi, 0,1,2,3,4,5,6,7,8,9,10,11,12,13,14,15);
}
__device__ __forceinline__ v8f zero8() {
  v8f z;
#pragma unroll
  for (int i = 0; i < 8; ++i) z[i] = 0.0f;
  return z;
}

// --- CDNA5 async global->LDS copy (ASYNCcnt-tracked DMA into LDS) ----------
__device__ __forceinline__ unsigned lds_off(const void* p) {
  // Flat LDS addresses carry the LDS byte offset in the low 32 bits
  // (see ISA section 10.2).
  return (unsigned)(unsigned long long)p;
}
__device__ __forceinline__ void async_b128(unsigned lds, const void* g) {
  asm volatile("global_load_async_to_lds_b128 %0, %1, off"
               :: "v"(lds), "v"(g) : "memory");
}
__device__ __forceinline__ void wait_async0() {
#if __has_builtin(__builtin_amdgcn_s_wait_asynccnt)
  __builtin_amdgcn_s_wait_asynccnt(0);
#else
  asm volatile("s_wait_asynccnt 0x0" ::: "memory");
#endif
}

// ---------------------------------------------------------------------------
// fp32 -> bf16 conversion (memory bound; 16B loads / 8B stores)
// ---------------------------------------------------------------------------
__global__ void cvt_f32_bf16(const float* __restrict__ in, bf16* __restrict__ out,
                             int n4) {
  int i = blockIdx.x * blockDim.x + threadIdx.x;
  if (i < n4) {
    float4 f = ((const float4*)in)[i];
    v4bf o;
    o[0] = (bf16)f.x; o[1] = (bf16)f.y; o[2] = (bf16)f.z; o[3] = (bf16)f.w;
    ((v4bf*)out)[i] = o;
  }
}

// ---------------------------------------------------------------------------
// GEMM core: C[M,N] = X[M,K] * W[N,K]^T, K=HIDDEN.
// Block tile 128x128, BK=32, double-buffered LDS filled by async-to-LDS DMA.
// 8 waves as 4(M)x2(N); each wave owns a 32x64 strip = 8 WMMA tiles.
// ---------------------------------------------------------------------------
__device__ __forceinline__ void stage_tiles_async(
    const bf16* __restrict__ X, const bf16* __restrict__ W,
    int rowBase, int colBase, int kk, unsigned sAoff, unsigned sBoff, int t) {
  // sA tile 128x32 bf16 = 512 x 16B chunks; 2 per thread
#pragma unroll
  for (int j = 0; j < 2; ++j) {
    int c = t * 2 + j;
    int r = c >> 2, e = (c & 3) * 8;
    async_b128(sAoff + (unsigned)(r * 32 + e) * 2u,
               &X[(size_t)(rowBase + r) * HIDDEN + kk + e]);
  }
  // sB tile 128x32 bf16 = 512 x 16B chunks; 2 per thread
#pragma unroll
  for (int j = 0; j < 2; ++j) {
    int c = t * 2 + j;
    int r = c >> 2, e = (c & 3) * 8;
    async_b128(sBoff + (unsigned)(r * 32 + e) * 2u,
               &W[(size_t)(colBase + r) * HIDDEN + kk + e]);
  }
}

struct GemmFrag {
  v8f acc[2][4];  // [mi][nt]
};

__device__ __forceinline__ void gemm_core(
    const bf16* __restrict__ X, const bf16* __restrict__ W,
    int rowBase, int colBase, GemmFrag& fr,
    bf16 (*sA)[128 * 32], bf16 (*sB)[128 * 32]) {
  const int t = threadIdx.x;
  const int w = t >> 5, l = t & 31, hl = l >> 4, ln = l & 15;
  const int wm = w >> 1, wn = w & 1;

#pragma unroll
  for (int mi = 0; mi < 2; ++mi)
#pragma unroll
    for (int nt = 0; nt < 4; ++nt) fr.acc[mi][nt] = zero8();

  unsigned aoff[2] = { lds_off(&sA[0][0]), lds_off(&sA[1][0]) };
  unsigned boff[2] = { lds_off(&sB[0][0]), lds_off(&sB[1][0]) };

  // Prologue: stage k-tile 0
  stage_tiles_async(X, W, rowBase, colBase, 0, aoff[0], boff[0], t);
  wait_async0();
  __syncthreads();

  for (int kk = 0; kk < HIDDEN; kk += 32) {
    const int cur = (kk >> 5) & 1;
    if (kk + 32 < HIDDEN)  // overlap: DMA next tile while WMMAs run
      stage_tiles_async(X, W, rowBase, colBase, kk + 32,
                        aoff[cur ^ 1], boff[cur ^ 1], t);

    const bf16* cA = sA[cur];
    const bf16* cB = sB[cur];
    // A fragments: row = wm*32 + mi*16 + ln, 16 contiguous K at hl*16
    v16bf af[2];
#pragma unroll
    for (int mi = 0; mi < 2; ++mi) {
      const bf16* p = &cA[(wm * 32 + mi * 16 + ln) * 32 + hl * 16];
      af[mi] = cat8(*(const v8bf*)p, *(const v8bf*)(p + 8));
    }
#pragma unroll
    for (int nt = 0; nt < 4; ++nt) {
      // B fragment: col = wn*64 + nt*16 + ln, 16 contiguous K at hl*16
      v16bf bfr = *(const v16bf*)&cB[(wn * 64 + nt * 16 + ln) * 32 + hl * 16];
      fr.acc[0][nt] = wmma_bf16(af[0], bfr, fr.acc[0][nt]);
      fr.acc[1][nt] = wmma_bf16(af[1], bfr, fr.acc[1][nt]);
    }
    wait_async0();     // next tile landed in LDS
    __syncthreads();   // all waves done reading cur / staging visible
  }
}

// QKV projection: scatter epilogue writes Q (pre-scaled), K row-major per head,
// V transposed [bh][D][S] (all bf16) for the attention kernel.
__global__ __launch_bounds__(256)
void gemm_qkv(const bf16* __restrict__ X, const bf16* __restrict__ W,
              bf16* __restrict__ Qb, bf16* __restrict__ Kb,
              bf16* __restrict__ Vt) {
  __shared__ bf16 sA[2][128 * 32];
  __shared__ bf16 sB[2][128 * 32];
  const int t = threadIdx.x;
  const int w = t >> 5, l = t & 31, hl = l >> 4, ln = l & 15;
  const int wm = w >> 1, wn = w & 1;
  const int rowBase = blockIdx.y * 128;
  const int colBase = blockIdx.x * 128;

  GemmFrag fr;
  gemm_core(X, W, rowBase, colBase, fr, sA, sB);

  // D layout: VGPR i -> row (i + 8*hl), col = lane&15
#pragma unroll
  for (int mi = 0; mi < 2; ++mi) {
#pragma unroll
    for (int nt = 0; nt < 4; ++nt) {
#pragma unroll
      for (int i = 0; i < 8; ++i) {
        int r = rowBase + wm * 32 + mi * 16 + hl * 8 + i;  // token index
        int o = colBase + wn * 64 + nt * 16 + ln;          // qkv column
        int which = o / HIDDEN;                            // 0=q 1=k 2=v
        int rem = o - which * HIDDEN;
        int h = rem / HDIM, d = rem - h * HDIM;
        int b = r >> 11, s = r & (SEQ - 1);
        int bh = b * HEADS + h;
        float v = fr.acc[mi][nt][i];
        if (which == 0)      Qb[(bh * SEQ + s) * HDIM + d] = (bf16)(v * QK_SCALE);
        else if (which == 1) Kb[(bh * SEQ + s) * HDIM + d] = (bf16)v;
        else                 Vt[(bh * HDIM + d) * SEQ + s] = (bf16)v;
      }
    }
  }
}

// Output projection: out[M,N] = X*W^T + bias, fp32 result.
__global__ __launch_bounds__(256)
void gemm_out(const bf16* __restrict__ X, const bf16* __restrict__ W,
              const float* __restrict__ bias, float* __restrict__ out) {
  __shared__ bf16 sA[2][128 * 32];
  __shared__ bf16 sB[2][128 * 32];
  const int t = threadIdx.x;
  const int w = t >> 5, l = t & 31, hl = l >> 4, ln = l & 15;
  const int wm = w >> 1, wn = w & 1;
  const int rowBase = blockIdx.y * 128;
  const int colBase = blockIdx.x * 128;

  GemmFrag fr;
  gemm_core(X, W, rowBase, colBase, fr, sA, sB);

#pragma unroll
  for (int mi = 0; mi < 2; ++mi) {
#pragma unroll
    for (int nt = 0; nt < 4; ++nt) {
#pragma unroll
      for (int i = 0; i < 8; ++i) {
        int r = rowBase + wm * 32 + mi * 16 + hl * 8 + i;
        int o = colBase + wn * 64 + nt * 16 + ln;
        out[r * HIDDEN + o] = fr.acc[mi][nt][i] + bias[o];
      }
    }
  }
}

// ---------------------------------------------------------------------------
// Flash attention: one wave per 16-query tile, online softmax over 32-key
// chunks. D=96 -> 3 K32 steps for Q*K^T, 6 N16 tiles for P*V.
// ---------------------------------------------------------------------------
__global__ __launch_bounds__(256, 2)
void attn(const bf16* __restrict__ Qb, const bf16* __restrict__ Kb,
          const bf16* __restrict__ Vt, bf16* __restrict__ O) {
  __shared__ bf16 sP[8][16 * 32];  // per-wave P staging (1 KB each)
  const int t = threadIdx.x;
  const int w = t >> 5, l = t & 31, hl = l >> 4, ln = l & 15;
  const int bh = blockIdx.y;
  const int b = bh / HEADS, h = bh - b * HEADS;
  const int s0 = (blockIdx.x * 8 + w) * 16;

  // Q fragments for this wave's 16 rows (Q pre-scaled by 1/sqrt(D))
  v16bf qf[3];
#pragma unroll
  for (int c = 0; c < 3; ++c) {
    const bf16* p = &Qb[(bh * SEQ + s0 + ln) * HDIM + c * 32 + hl * 16];
    qf[c] = cat8(*(const v8bf*)p, *(const v8bf*)(p + 8));
  }

  v8f oacc[6];
#pragma unroll
  for (int dt = 0; dt < 6; ++dt) oacc[dt] = zero8();
  float mi[8], li[8];
#pragma unroll
  for (int i = 0; i < 8; ++i) { mi[i] = -1e30f; li[i] = 0.0f; }

  for (int kb = 0; kb < SEQ; kb += 32) {
    // scores for two 16-key tiles
    v8f sc0 = zero8(), sc1 = zero8();
#pragma unroll
    for (int c = 0; c < 3; ++c) {
      v16bf k0 = *(const v16bf*)&Kb[(bh * SEQ + kb + ln) * HDIM + c * 32 + hl * 16];
      v16bf k1 = *(const v16bf*)&Kb[(bh * SEQ + kb + 16 + ln) * HDIM + c * 32 + hl * 16];
      sc0 = wmma_bf16(qf[c], k0, sc0);
      sc1 = wmma_bf16(qf[c], k1, sc1);
    }
    // online softmax per row (rows spread across 16-lane groups)
    float p0[8], p1[8], rs[8];
#pragma unroll
    for (int i = 0; i < 8; ++i) {
      float mx = fmaxf(sc0[i], sc1[i]);
      mx = fmaxf(mx, __shfl_xor(mx, 1, 32));
      mx = fmaxf(mx, __shfl_xor(mx, 2, 32));
      mx = fmaxf(mx, __shfl_xor(mx, 4, 32));
      mx = fmaxf(mx, __shfl_xor(mx, 8, 32));
      float mn = fmaxf(mi[i], mx);
      float scale = __expf(mi[i] - mn);
      p0[i] = __expf(sc0[i] - mn);
      p1[i] = __expf(sc1[i] - mn);
      float ps = p0[i] + p1[i];
      ps += __shfl_xor(ps, 1, 32);
      ps += __shfl_xor(ps, 2, 32);
      ps += __shfl_xor(ps, 4, 32);
      ps += __shfl_xor(ps, 8, 32);
      li[i] = li[i] * scale + ps;
      mi[i] = mn;
      rs[i] = scale;
    }
#pragma unroll
    for (int dt = 0; dt < 6; ++dt)
#pragma unroll
      for (int i = 0; i < 8; ++i) oacc[dt][i] *= rs[i];

    // Re-layout P (D-layout) into A fragments via per-wave LDS.
#pragma unroll
    for (int i = 0; i < 8; ++i) {
      int m = i + hl * 8;
      sP[w][m * 32 + ln]      = (bf16)p0[i];
      sP[w][m * 32 + 16 + ln] = (bf16)p1[i];
    }
    __builtin_amdgcn_wave_barrier();  // wave-lockstep; DS in-order per wave
    v16bf pa = cat8(*(const v8bf*)&sP[w][ln * 32 + hl * 16],
                    *(const v8bf*)&sP[w][ln * 32 + hl * 16 + 8]);
    __builtin_amdgcn_wave_barrier();

    // out += P(16x32) * V(32x96) ; Vt is [bh][D][S] so B loads are contiguous
#pragma unroll
    for (int dt = 0; dt < 6; ++dt) {
      v16bf vf = *(const v16bf*)&Vt[(bh * HDIM + dt * 16 + ln) * SEQ + kb + hl * 16];
      oacc[dt] = wmma_bf16(pa, vf, oacc[dt]);
    }
  }

  float inv[8];
#pragma unroll
  for (int i = 0; i < 8; ++i) inv[i] = 1.0f / li[i];
#pragma unroll
  for (int dt = 0; dt < 6; ++dt)
#pragma unroll
    for (int i = 0; i < 8; ++i) {
      int r = s0 + i + hl * 8;
      int col = h * HDIM + dt * 16 + ln;
      O[(b * SEQ + r) * HIDDEN + col] = (bf16)(oacc[dt][i] * inv[i]);
    }
}

// ---------------------------------------------------------------------------
// Host launcher
// ---------------------------------------------------------------------------
extern "C" void kernel_launch(void* const* d_in, const int* in_sizes, int n_in,
                              void* d_out, int out_size, void* d_ws, size_t ws_size,
                              hipStream_t stream) {
  const float* x     = (const float*)d_in[0];
  const float* w_qkv = (const float*)d_in[1];
  const float* w_o   = (const float*)d_in[2];
  const float* b_o   = (const float*)d_in[3];

  char* ws = (char*)d_ws;
  // bf16 workspace regions (bytes, all 256B aligned)
  bf16* xb   = (bf16*)(ws + 0);          // x            : 25,165,824
  bf16* wqb  = (bf16*)(ws + 25165824);   // w_qkv        : 56,623,104
  bf16* wob  = (bf16*)(ws + 81788928);   // w_o          : 18,874,368
  bf16* qb   = (bf16*)(ws + 100663296);  // Q [bh][S][D] : 25,165,824
  bf16* kbuf = (bf16*)(ws + 125829120);  // K [bh][S][D] : 25,165,824
  bf16* vtb  = (bf16*)(ws + 150994944);  // V [bh][D][S] : 25,165,824
  bf16* ob   = (bf16*)(ws + 176160768);  // attn out     : 25,165,824

  // 1) fp32 -> bf16 conversions
  {
    int n4 = (ROWS * HIDDEN) / 4;
    cvt_f32_bf16<<<(n4 + 255) / 256, 256, 0, stream>>>(x, xb, n4);
  }
  {
    int n4 = (NQKV * HIDDEN) / 4;
    cvt_f32_bf16<<<(n4 + 255) / 256, 256, 0, stream>>>(w_qkv, wqb, n4);
  }
  {
    int n4 = (HIDDEN * HIDDEN) / 4;
    cvt_f32_bf16<<<(n4 + 255) / 256, 256, 0, stream>>>(w_o, wob, n4);
  }

  // 2) QKV projection GEMM (scatters per-head Q/K/V^T)
  gemm_qkv<<<dim3(NQKV / 128, ROWS / 128), 256, 0, stream>>>(xb, wqb, qb, kbuf, vtb);

  // 3) Flash attention: 128 q-tiles per (b,h) / 8 waves per block
  attn<<<dim3(16, NBH), 256, 0, stream>>>(qb, kbuf, vtb, ob);

  // 4) Output projection GEMM + bias -> fp32
  gemm_out<<<dim3(HIDDEN / 128, ROWS / 128), 256, 0, stream>>>(ob, wob, b_o,
                                                              (float*)d_out);
}
